// MultiHeadAttention_78469052498170
// MI455X (gfx1250) — matrix operand order repaired and checked by reading, not compiled, not run on hardware
//
#include <hip/hip_runtime.h>
#include <hip/hip_bf16.h>
#include <stdint.h>

// ---------------------------------------------------------------------------
// MHA for MI455X (gfx1250): bf16 WMMA (16x16x32) GEMMs + wave-level flash attn
// B=4, T=2048, C=1024, H=16, hd=64
// ---------------------------------------------------------------------------

typedef __attribute__((ext_vector_type(16))) __bf16 v16bf;
typedef __attribute__((ext_vector_type(8)))  float  v8f;

#define DEVINL __device__ __forceinline__

static constexpr int B_  = 4;
static constexpr int T_  = 2048;
static constexpr int C_  = 1024;
static constexpr int H_  = 16;
static constexpr int HD_ = 64;
static constexpr int M_  = B_ * T_;   // 8192 rows for all GEMMs

// Native bf16 convert (RTNE) -> v_cvt_*bf16_f32 on gfx1250.
DEVINL unsigned short f2bf(float f) {
  return __builtin_bit_cast(unsigned short, (__bf16)f);
}

union Frag {
  v16bf v;
  uint4 q[2];
  unsigned short us[16];
};

// A-matrix fragment (16 rows x 32 k, bf16), rows of `stride` elements.
// ISA 16-bit A layout: lane<16 -> row=lane, k {0..7,16..23}; lane>=16 -> k {8..15,24..31}.
DEVINL v16bf frag_a(const __bf16* base, int stride) {
  int l = (int)(threadIdx.x & 31);
  const __bf16* p = base + (l & 15) * stride + ((l < 16) ? 0 : 8);
  Frag f;
  f.q[0] = *(const uint4*)(p);
  f.q[1] = *(const uint4*)(p + 16);
  return f.v;
}

// B-matrix fragment (32 k x 16 n, bf16) gathered from n-major storage:
// base[n*stride + k].  ISA B layout: lane<16 -> n=lane, k {0..7,16..23}; lane>=16 -> k {8..15,24..31}.
DEVINL v16bf frag_b(const __bf16* base, int stride) {
  int l = (int)(threadIdx.x & 31);
  const __bf16* p = base + (l & 15) * stride + ((l < 16) ? 0 : 8);
  Frag f;
  f.q[0] = *(const uint4*)(p);
  f.q[1] = *(const uint4*)(p + 16);
  return f.v;
}

// Constant all-ones B fragment: rowsum(P) = P @ ones via the matrix core.
DEVINL v16bf ones_frag() {
  Frag f;
#pragma unroll
  for (int i = 0; i < 16; ++i) f.us[i] = 0x3F80;  // bf16 1.0
  return f.v;
}

DEVINL v8f wmma_bf16(v16bf a, v16bf b, v8f c) {
  return __builtin_amdgcn_wmma_f32_16x16x32_bf16(
      /*neg_a=*/false, a, /*neg_b=*/false, b,
      /*c_mod=*/(short)0, c, /*reuse_a=*/false, /*reuse_b=*/false);
}

// ---------------------------------------------------------------------------
// GEMM: out = A(M x C) @ W(C x C) + bias, with bf16 compute / f32 accumulate.
// AMODE: 0 = A fp32, 1 = A bf16.
// OMODE: 0 = bf16 out in (B,H,T,hd); 1 = bf16 out in (B,H,hd,T); 2 = fp32 row-major.
// Block: 256 threads (8 waves), 128x64 block tile, 32x32 per wave, K-chunk 64.
// ---------------------------------------------------------------------------
template <int AMODE, int OMODE>
__global__ __launch_bounds__(256) void mha_gemm_kernel(const void* __restrict__ Av,
                                                       const float* __restrict__ W,
                                                       const float* __restrict__ bias,
                                                       void* __restrict__ outv) {
  __shared__ __align__(16) __bf16 lds_a[128 * 64];   // A tile, row-major (16 KB)
  __shared__ __align__(16) __bf16 lds_wt[64 * 64];   // W tile, n-major   (8 KB)

  const int tid = (int)threadIdx.x;
  const int bm  = (int)blockIdx.x * 128;
  const int bn  = (int)blockIdx.y * 64;
  const int wid = tid >> 5;
  const int l   = tid & 31;
  const int waveM = (wid & 3) * 32;
  const int waveN = (wid >> 2) * 32;

  v8f acc[2][2] = {};

  for (int k0 = 0; k0 < C_; k0 += 64) {
    // ---- stage A tile: 128 x 64 -> bf16 LDS (32 elements/thread) ----
    {
      const int row = tid >> 1;
      const int cg  = (tid & 1) * 32;
      const size_t g = (size_t)(bm + row) * C_ + k0 + cg;
      Frag t[2];
      if (AMODE == 0) {
        const float* A = (const float*)Av;
#pragma unroll
        for (int v = 0; v < 2; ++v) {
#pragma unroll
          for (int i = 0; i < 16; i += 4) {
            float4 f = *(const float4*)(A + g + v * 16 + i);
            t[v].us[i + 0] = f2bf(f.x); t[v].us[i + 1] = f2bf(f.y);
            t[v].us[i + 2] = f2bf(f.z); t[v].us[i + 3] = f2bf(f.w);
          }
        }
      } else {
        const __bf16* A = (const __bf16*)Av;
        t[0].q[0] = *(const uint4*)(A + g);
        t[0].q[1] = *(const uint4*)(A + g + 8);
        t[1].q[0] = *(const uint4*)(A + g + 16);
        t[1].q[1] = *(const uint4*)(A + g + 24);
      }
      *(uint4*)(lds_a + row * 64 + cg)      = t[0].q[0];
      *(uint4*)(lds_a + row * 64 + cg + 8)  = t[0].q[1];
      *(uint4*)(lds_a + row * 64 + cg + 16) = t[1].q[0];
      *(uint4*)(lds_a + row * 64 + cg + 24) = t[1].q[1];
    }
    // ---- stage W tile: 64 x 64 -> bf16 LDS, transposed to n-major ----
    {
      const int k  = tid >> 2;           // 0..63
      const int ng = (tid & 3) * 16;     // 0,16,32,48
      const float* wp = W + (size_t)(k0 + k) * C_ + bn + ng;
      unsigned short w16[16];
#pragma unroll
      for (int i = 0; i < 16; i += 4) {
        float4 f = *(const float4*)(wp + i);
        w16[i + 0] = f2bf(f.x); w16[i + 1] = f2bf(f.y);
        w16[i + 2] = f2bf(f.z); w16[i + 3] = f2bf(f.w);
      }
      unsigned short* wt = (unsigned short*)lds_wt;
#pragma unroll
      for (int i = 0; i < 16; ++i) wt[(ng + i) * 64 + k] = w16[i];
    }
    __syncthreads();

#pragma unroll
    for (int kh = 0; kh < 2; ++kh) {     // two 32-wide K sub-chunks
      v16bf af[2], bf[2];
      af[0] = frag_a(lds_a + (waveM + 0)  * 64 + kh * 32, 64);
      af[1] = frag_a(lds_a + (waveM + 16) * 64 + kh * 32, 64);
      bf[0] = frag_b(lds_wt + (waveN + 0)  * 64 + kh * 32, 64);
      bf[1] = frag_b(lds_wt + (waveN + 16) * 64 + kh * 32, 64);
#pragma unroll
      for (int mt = 0; mt < 2; ++mt)
#pragma unroll
        for (int nt = 0; nt < 2; ++nt)
          acc[mt][nt] = wmma_bf16(af[mt], bf[nt], acc[mt][nt]);
    }
    __syncthreads();
  }

  // ---- epilogue: bias + layout-specific store ----
  const int half8 = (l >= 16) ? 8 : 0;
  const int nl = l & 15;
#pragma unroll
  for (int mt = 0; mt < 2; ++mt) {
#pragma unroll
    for (int nt = 0; nt < 2; ++nt) {
#pragma unroll
      for (int r = 0; r < 8; ++r) {
        const int gm = bm + waveM + mt * 16 + r + half8;
        const int gn = bn + waveN + nt * 16 + nl;
        const float val = acc[mt][nt][r] + bias[gn];
        if (OMODE == 2) {
          ((float*)outv)[(size_t)gm * C_ + gn] = val;
        } else {
          const int b = gm >> 11, t = gm & (T_ - 1);
          const int h = gn >> 6,  d = gn & (HD_ - 1);
          size_t idx;
          if (OMODE == 0) idx = (((size_t)(b * H_ + h)) * T_ + t) * HD_ + d;  // (B,H,T,hd)
          else            idx = (((size_t)(b * H_ + h)) * HD_ + d) * T_ + t;  // (B,H,hd,T)
          ((unsigned short*)outv)[idx] = f2bf(val);
        }
      }
    }
  }
}

// ---------------------------------------------------------------------------
// Flash attention: one wave per 16-row q-tile. Q/K in (B,H,T,hd) bf16,
// V transposed (B,H,hd,T) bf16. Online softmax (base-2), padding mask
// (branchless); causal compare only in the single diagonal chunk.
// Row sums via WMMA against all-ones B. Output bf16 (B,T,C).
// ---------------------------------------------------------------------------
__global__ __launch_bounds__(128) void mha_flash_kernel(const __bf16* __restrict__ Q,
                                                        const __bf16* __restrict__ K,
                                                        const __bf16* __restrict__ Vt,
                                                        const unsigned char* __restrict__ mask,
                                                        unsigned short* __restrict__ attnout) {
  __shared__ __align__(16) __bf16 pbuf[4][16 * 32];  // one 16x32 P tile per wave

  const int wid = (int)(threadIdx.x >> 5);
  const int l   = (int)(threadIdx.x & 31);
  const int gwave = (int)blockIdx.x * 4 + wid;
  const int qt = gwave & 127;            // T/16 = 128 q-tiles per (b,h)
  const int bh = gwave >> 7;             // 0..63
  const int b = bh >> 4, h = bh & 15;
  const int qb = qt * 16;

  const __bf16* Qbh = Q  + (size_t)bh * T_ * HD_;
  const __bf16* Kbh = K  + (size_t)bh * T_ * HD_;
  const __bf16* Vbh = Vt + (size_t)bh * HD_ * T_;
  const unsigned char* mb = mask + (size_t)b * T_ * T_;
  __bf16* pb = pbuf[wid];

  // Q fragments for this q-tile (hd chunks 0..31 and 32..63), reused all loop.
  const v16bf qf0 = frag_a(Qbh + (size_t)qb * HD_ + 0,  HD_);
  const v16bf qf1 = frag_a(Qbh + (size_t)qb * HD_ + 32, HD_);
  const v16bf onef = ones_frag();

  v8f o[4] = {};
  float mrow[8], lrow[8];
#pragma unroll
  for (int r = 0; r < 8; ++r) { mrow[r] = -3.0e38f; lrow[r] = 0.0f; }

  const int half8 = (l >= 16) ? 8 : 0;
  const int nl = l & 15;
  // scores kept in base-2 domain: s * (1/sqrt(hd)) * log2(e)
  const float scale2 = 0.125f * 1.44269504088896340736f;

  // One 32-wide kv chunk; `causal` constant-folds at each call site.
  auto chunk = [&](int kv0, bool causal) {
    // ---- scores: two 16x16 subtiles, contraction over hd=64 (2 x K32) ----
    v8f s[2];
#pragma unroll
    for (int st = 0; st < 2; ++st) {
      const __bf16* kb = Kbh + (size_t)(kv0 + st * 16) * HD_;
      v8f a = {};
      a = wmma_bf16(qf0, frag_b(kb + 0,  HD_), a);
      a = wmma_bf16(qf1, frag_b(kb + 32, HD_), a);
      s[st] = a;
    }

    // ---- padding mask bytes: unconditional clustered loads (no branches) ----
    unsigned int mv[2][8];
#pragma unroll
    for (int st = 0; st < 2; ++st)
#pragma unroll
      for (int r = 0; r < 8; ++r)
        mv[st][r] = mb[(size_t)(qb + r + half8) * T_ + kv0 + st * 16 + nl];

    // ---- scale + mask, branchless; causal compare only on diagonal chunk ----
#pragma unroll
    for (int st = 0; st < 2; ++st) {
      const int kv = kv0 + st * 16 + nl;
#pragma unroll
      for (int r = 0; r < 8; ++r) {
        unsigned int ok = (unsigned int)(mv[st][r] != 0);
        if (causal) ok &= (unsigned int)(kv <= qb + r + half8);
        s[st][r] = ok ? s[st][r] * scale2 : -3.0e38f;
      }
    }

    // ---- online softmax: row max via 16-lane butterflies ----
    float alpha[8];
#pragma unroll
    for (int r = 0; r < 8; ++r) {
      float v = fmaxf(s[0][r], s[1][r]);
#pragma unroll
      for (int off = 1; off < 16; off <<= 1) v = fmaxf(v, __shfl_xor(v, off, 32));
      const float mi = fmaxf(mrow[r], v);
      alpha[r] = exp2f(mrow[r] - mi);
      mrow[r] = mi;
    }

    // ---- P = exp2(s - m) straight into LDS as bf16 ----
    unsigned short* pw = (unsigned short*)pb;
#pragma unroll
    for (int r = 0; r < 8; ++r) {
      pw[(r + half8) * 32 + nl]      = f2bf(exp2f(s[0][r] - mrow[r]));
      pw[(r + half8) * 32 + 16 + nl] = f2bf(exp2f(s[1][r] - mrow[r]));
    }

    // ---- rescale O while the stores land ----
#pragma unroll
    for (int nt = 0; nt < 4; ++nt)
#pragma unroll
      for (int r = 0; r < 8; ++r) o[nt][r] *= alpha[r];

    asm volatile("s_wait_dscnt 0" ::: "memory");   // wave-internal LDS RAW
    const v16bf pf = frag_a(pb, 32);

    // ---- row sums via matrix core: P @ ones (C layout row = r+half8) ----
    v8f srow = wmma_bf16(pf, onef, (v8f){});
#pragma unroll
    for (int r = 0; r < 8; ++r) lrow[r] = lrow[r] * alpha[r] + srow[r];

    // ---- O += P @ V ----
#pragma unroll
    for (int nt = 0; nt < 4; ++nt) {
      const v16bf vf = frag_b(Vbh + (size_t)(nt * 16) * T_ + kv0, T_);
      o[nt] = wmma_bf16(pf, vf, o[nt]);
    }
  };

  const int ni = qb >> 5;                 // interior chunks: kv0+31 < qb <= qrow
  for (int kc = 0; kc < ni; ++kc) chunk(kc * 32, false);
  chunk(ni * 32, true);                   // single diagonal chunk

  // ---- normalize + store to (B,T,C) bf16 ----
#pragma unroll
  for (int r = 0; r < 8; ++r) lrow[r] = 1.0f / lrow[r];
#pragma unroll
  for (int nt = 0; nt < 4; ++nt) {
#pragma unroll
    for (int r = 0; r < 8; ++r) {
      const int qrow = qb + r + half8;
      const int col = h * HD_ + nt * 16 + nl;
      attnout[((size_t)b * T_ + qrow) * C_ + col] = f2bf(o[nt][r] * lrow[r]);
    }
  }
}

// ---------------------------------------------------------------------------
// Launch: Q/K/V projections -> flash attention -> output projection
// inputs: k,q,v,mask,Wk,bk,Wq,bq,Wv,bv,Wo,bo
// ---------------------------------------------------------------------------
extern "C" void kernel_launch(void* const* d_in, const int* in_sizes, int n_in,
                              void* d_out, int out_size, void* d_ws, size_t ws_size,
                              hipStream_t stream) {
  (void)in_sizes; (void)n_in; (void)out_size; (void)ws_size;

  const float* k  = (const float*)d_in[0];
  const float* q  = (const float*)d_in[1];
  const float* v  = (const float*)d_in[2];
  const unsigned char* mask = (const unsigned char*)d_in[3];
  const float* Wk = (const float*)d_in[4];
  const float* bk = (const float*)d_in[5];
  const float* Wq = (const float*)d_in[6];
  const float* bq = (const float*)d_in[7];
  const float* Wv = (const float*)d_in[8];
  const float* bv = (const float*)d_in[9];
  const float* Wo = (const float*)d_in[10];
  const float* bo = (const float*)d_in[11];

  char* ws = (char*)d_ws;
  const size_t SEG = (size_t)B_ * H_ * T_ * HD_ * 2;  // 16 MiB bf16 per tensor
  __bf16* Qp = (__bf16*)(ws + 0 * SEG);
  __bf16* Kp = (__bf16*)(ws + 1 * SEG);
  __bf16* Vt = (__bf16*)(ws + 2 * SEG);
  __bf16* AO = (__bf16*)(ws + 3 * SEG);

  dim3 gblk(256);
  dim3 ggrid(M_ / 128, C_ / 64);   // 64 x 16

  mha_gemm_kernel<0, 0><<<ggrid, gblk, 0, stream>>>((const void*)q, Wq, bq, (void*)Qp);
  mha_gemm_kernel<0, 0><<<ggrid, gblk, 0, stream>>>((const void*)k, Wk, bk, (void*)Kp);
  mha_gemm_kernel<0, 1><<<ggrid, gblk, 0, stream>>>((const void*)v, Wv, bv, (void*)Vt);

  const int nwaves = B_ * H_ * (T_ / 16);          // 8192 q-tile waves
  mha_flash_kernel<<<nwaves / 4, 128, 0, stream>>>(Qp, Kp, Vt, mask,
                                                   (unsigned short*)AO);

  mha_gemm_kernel<1, 2><<<ggrid, gblk, 0, stream>>>((const void*)AO, Wo, bo, d_out);
}
